// GraphTransformer_71777493451184
// MI455X (gfx1250) — matrix operand order repaired and checked by reading, not compiled, not run on hardware
//
#include <hip/hip_runtime.h>
#include <hip/hip_bf16.h>

// ---------------------------------------------------------------------------
// GraphTransformer for MI455X (gfx1250, wave32).
//  - Dense GEMMs: v_wmma_f32_16x16x4_f32 (f32-exact matrix path),
//    template-specialized so the K-loop is branch-free.
//  - MHA (dominant, 34 GFLOP): two-pass flash attention with
//    v_wmma_f32_16x16x32_bf16 (bf16 inputs, f32 accumulate)
//  - Graph scatter softmax: float atomic max/add (L2-resident accumulators)
// ---------------------------------------------------------------------------

typedef __attribute__((ext_vector_type(2)))  float   v2f;
typedef __attribute__((ext_vector_type(8)))  float   v8f;
typedef __attribute__((ext_vector_type(16))) __bf16  v16bf;

#define N_NODES 8192
#define N_EDGES 262144
#define E_TOT   (N_EDGES + N_NODES)   // + self loops
#define HD      512                    // 4 heads * 128
#define NEG_SLOPE 0.2f

// ---------------- small helpers ----------------

__device__ __forceinline__ void atomicMaxF(float* addr, float val) {
  if (val >= 0.f) atomicMax((int*)addr, __float_as_int(val));
  else            atomicMin((unsigned int*)addr, __float_as_uint(val));
}

__global__ void fill_kernel(float* __restrict__ p, float v, int n) {
  int i = blockIdx.x * 256 + threadIdx.x;
  if (i < n) p[i] = v;
}

// ---------------- generic f32 WMMA GEMM (compile-time specialized) ---------
// C[M,Nc] = (RELU?)(A[M,K] @ B + bias), B is [K,Nc] or (TRANSB) [Nc,K].
// One wave computes a 16x64 strip via 4 accumulators; block = 8 waves
// -> 128 rows x 64 cols per block. M%128==0, Nc%64==0, K%4==0 (all hold here).
template <int TRANSB, int RELU, int BIAS>
__global__ __launch_bounds__(256)
void gemm_f32_kernel(const float* __restrict__ A,
                     const float* __restrict__ B,
                     const float* __restrict__ bias,
                     float* __restrict__ C,
                     int M, int Nc, int K) {
  const int wave = threadIdx.x >> 5;
  const int lane = threadIdx.x & 31;
  const int half = lane >> 4;        // K-half select
  const int n    = lane & 15;        // row (A) / col (B,C) within tile
  const int row0 = blockIdx.y * 128 + wave * 16;
  const int col0 = blockIdx.x * 64;

  // A 16x4 layout: lanes0-15 K={k0,k0+1}, lanes16-31 K={k0+2,k0+3}
  const float* ap = A + (size_t)(row0 + n) * K + 2 * half;

  const float* bp0;
  const float* bp1;
  const float* bp2;
  const float* bp3;
  if (TRANSB) {                       // B[Nc,K]: col-major access, contiguous K
    bp0 = B + (size_t)(col0 + 0 * 16 + n) * K + 2 * half;
    bp1 = B + (size_t)(col0 + 1 * 16 + n) * K + 2 * half;
    bp2 = B + (size_t)(col0 + 2 * 16 + n) * K + 2 * half;
    bp3 = B + (size_t)(col0 + 3 * 16 + n) * K + 2 * half;
  } else {                            // B[K,Nc]: strided by Nc between K rows
    bp0 = B + (size_t)(2 * half) * Nc + col0 + 0 * 16 + n;
    bp1 = bp0 + 16; bp2 = bp0 + 32; bp3 = bp0 + 48;
  }

  v8f acc0 = {}, acc1 = {}, acc2 = {}, acc3 = {};
#pragma unroll 2
  for (int k0 = 0; k0 < K; k0 += 4) {
    v2f a; a[0] = ap[0]; a[1] = ap[1]; ap += 4;
    v2f b0, b1, b2, b3;
    if (TRANSB) {
      b0[0] = bp0[0]; b0[1] = bp0[1]; bp0 += 4;
      b1[0] = bp1[0]; b1[1] = bp1[1]; bp1 += 4;
      b2[0] = bp2[0]; b2[1] = bp2[1]; bp2 += 4;
      b3[0] = bp3[0]; b3[1] = bp3[1]; bp3 += 4;
    } else {
      b0[0] = bp0[0]; b0[1] = bp0[Nc];
      b1[0] = bp1[0]; b1[1] = bp1[Nc];
      b2[0] = bp2[0]; b2[1] = bp2[Nc];
      b3[0] = bp3[0]; b3[1] = bp3[Nc];
      const size_t step = (size_t)4 * Nc;
      bp0 += step; bp1 += step; bp2 += step; bp3 += step;
    }
    acc0 = __builtin_amdgcn_wmma_f32_16x16x4_f32(false, a, false, b0, (short)0, acc0, false, false);
    acc1 = __builtin_amdgcn_wmma_f32_16x16x4_f32(false, a, false, b1, (short)0, acc1, false, false);
    acc2 = __builtin_amdgcn_wmma_f32_16x16x4_f32(false, a, false, b2, (short)0, acc2, false, false);
    acc3 = __builtin_amdgcn_wmma_f32_16x16x4_f32(false, a, false, b3, (short)0, acc3, false, false);
  }

  v8f* accs[4] = {&acc0, &acc1, &acc2, &acc3};
#pragma unroll
  for (int t = 0; t < 4; ++t) {
    const int col = col0 + t * 16 + n;
    const float bv = BIAS ? bias[col] : 0.f;
    const v8f av = *accs[t];
#pragma unroll
    for (int i = 0; i < 8; ++i) {
      const int row = row0 + i + 8 * half;   // C/D layout
      float v = av[i] + bv;
      if (RELU) v = fmaxf(v, 0.f);
      C[(size_t)row * Nc + col] = v;
    }
  }
}

// ---------------- GAT pieces ----------------

// al_s[n,h] = sum_c hl[n,h*128+c]*att_src[h,c] ; al_d likewise. Wave per (n,h).
__global__ __launch_bounds__(256)
void gat_alpha_kernel(const float* __restrict__ hl,
                      const float* __restrict__ asrc,
                      const float* __restrict__ adst,
                      float* __restrict__ als,
                      float* __restrict__ ald) {
  const int wave = blockIdx.x * 8 + (threadIdx.x >> 5);
  const int lane = threadIdx.x & 31;
  const int node = wave >> 2, h = wave & 3;
  const float* hp = hl + (size_t)node * HD + h * 128;
  const float* ap = asrc + h * 128;
  const float* dp = adst + h * 128;
  float ss = 0.f, sd = 0.f;
#pragma unroll
  for (int c0 = 0; c0 < 128; c0 += 32) {
    const int c = c0 + lane;
    const float v = hp[c];
    ss += v * ap[c]; sd += v * dp[c];
  }
  for (int off = 16; off >= 1; off >>= 1) {
    ss += __shfl_xor(ss, off, 32);
    sd += __shfl_xor(sd, off, 32);
  }
  if (lane == 0) { als[node * 4 + h] = ss; ald[node * 4 + h] = sd; }
}

__device__ __forceinline__ void edge_sd(const int* ei, int e, int& s, int& d) {
  if (e < N_EDGES) { s = ei[e]; d = ei[N_EDGES + e]; }
  else             { s = e - N_EDGES; d = s; }           // self loop
}

// leaky-relu edge scores + segment max (over dst)
__global__ __launch_bounds__(256)
void edge_score_kernel(const int* __restrict__ ei,
                       const float* __restrict__ als,
                       const float* __restrict__ ald,
                       float* __restrict__ ev,
                       float* __restrict__ smax) {
  const int idx = blockIdx.x * 256 + threadIdx.x;
  if (idx >= E_TOT * 4) return;
  const int e = idx >> 2, h = idx & 3;
  int s, d; edge_sd(ei, e, s, d);
  float v = als[s * 4 + h] + ald[d * 4 + h];
  v = v > 0.f ? v : NEG_SLOPE * v;
  ev[idx] = v;
  atomicMaxF(&smax[d * 4 + h], v);
}

// exp(e - segmax) + segment sum
__global__ __launch_bounds__(256)
void edge_exp_kernel(const int* __restrict__ ei,
                     float* __restrict__ ev,
                     const float* __restrict__ smax,
                     float* __restrict__ ssum) {
  const int idx = blockIdx.x * 256 + threadIdx.x;
  if (idx >= E_TOT * 4) return;
  const int e = idx >> 2, h = idx & 3;
  int s, d; edge_sd(ei, e, s, d); (void)s;
  const float ex = __expf(ev[idx] - smax[d * 4 + h]);
  ev[idx] = ex;
  atomicAdd(&ssum[d * 4 + h], ex);
}

// agg[d,h,:] += alpha * hl[s,h,:]  — wave per (edge,head), lane covers 4 chans
__global__ __launch_bounds__(256)
void edge_agg_kernel(const int* __restrict__ ei,
                     const float* __restrict__ ev,
                     const float* __restrict__ ssum,
                     const float* __restrict__ hl,
                     float* __restrict__ agg) {
  const int w = blockIdx.x * 8 + (threadIdx.x >> 5);
  if (w >= E_TOT * 4) return;
  const int lane = threadIdx.x & 31;
  const int e = w >> 2, h = w & 3;
  int s, d; edge_sd(ei, e, s, d);
  const float alpha = ev[w] / (ssum[d * 4 + h] + 1e-16f);
  const float4 hv = *(const float4*)(hl + (size_t)s * HD + h * 128 + lane * 4);
  float* op = agg + (size_t)d * HD + h * 128 + lane * 4;
  atomicAdd(op + 0, alpha * hv.x);
  atomicAdd(op + 1, alpha * hv.y);
  atomicAdd(op + 2, alpha * hv.z);
  atomicAdd(op + 3, alpha * hv.w);
}

__global__ __launch_bounds__(256)
void bias_relu_kernel(const float* __restrict__ agg,
                      const float* __restrict__ b,
                      float* __restrict__ out, int total) {
  const int i = blockIdx.x * 256 + threadIdx.x;
  if (i < total) out[i] = fmaxf(agg[i] + b[i & (HD - 1)], 0.f);
}

// ---------------- flash attention (two passes, bf16 WMMA) ----------------
// qkv row layout: [q(128) | k(128) | v(128)], head h uses dims h*32..h*32+31.
// One wave handles 16 queries of one head; waves = 4 * (N/16).

__device__ __forceinline__ v16bf load_a_bf16(const float* qrow, int half, float scale) {
  // A 16x32 bf16 operand in this code's k order (B matches): lane holds K = 16h..16h+15 contiguous
  const float* cA = qrow + 16 * half;
  const float* cB = qrow + 16 * half + 8;
  v16bf r;
#pragma unroll
  for (int i = 0; i < 8; ++i) {
    r[i]     = (__bf16)(cA[i] * scale);
    r[8 + i] = (__bf16)(cB[i] * scale);
  }
  return r;
}

__global__ __launch_bounds__(256)
void attn_pass1_kernel(const float* __restrict__ qkv,
                       float* __restrict__ mrow,
                       float* __restrict__ lrow, int N) {
  const int wave = (blockIdx.x << 3) + (threadIdx.x >> 5);
  const int lane = threadIdx.x & 31;
  const int half = lane >> 4;
  const int n    = lane & 15;
  const int qpt  = N >> 4;
  const int h    = wave / qpt;
  const int q0   = (wave % qpt) << 4;
  const float scale = 0.17677669529663689f;  // 1/sqrt(32)

  const v16bf aq = load_a_bf16(qkv + (size_t)(q0 + n) * 384 + h * 32, half, scale);

  float mloc[8], lloc[8];
#pragma unroll
  for (int i = 0; i < 8; ++i) { mloc[i] = -3.0e38f; lloc[i] = 0.f; }

  for (int j0 = 0; j0 < N; j0 += 16) {
    // B 32x16 operand in this code's k order (matches A): lane col = key j0+n, K(dim) = 16*half + 0..15 contiguous
    const float* krow = qkv + (size_t)(j0 + n) * 384 + 128 + h * 32 + 16 * half;
    v16bf bk;
#pragma unroll
    for (int i = 0; i < 16; ++i) bk[i] = (__bf16)krow[i];
    v8f s = {};
    s = __builtin_amdgcn_wmma_f32_16x16x32_bf16(false, aq, false, bk, (short)0, s,
                                                false, false);
#pragma unroll
    for (int i = 0; i < 8; ++i) {
      const float sv = s[i];
      const float nm = fmaxf(mloc[i], sv);
      lloc[i] = lloc[i] * __expf(mloc[i] - nm) + __expf(sv - nm);
      mloc[i] = nm;
    }
  }
  // merge (m,l) across the 16 lanes of each half (same 8 rows, disjoint cols)
#pragma unroll
  for (int i = 0; i < 8; ++i) {
    for (int off = 8; off >= 1; off >>= 1) {
      const float mo = __shfl_xor(mloc[i], off, 32);
      const float lo = __shfl_xor(lloc[i], off, 32);
      const float nm = fmaxf(mloc[i], mo);
      lloc[i] = lloc[i] * __expf(mloc[i] - nm) + lo * __expf(mo - nm);
      mloc[i] = nm;
    }
    if (n == 0) {
      const int row = q0 + i + 8 * half;
      mrow[h * N + row] = mloc[i];
      lrow[h * N + row] = lloc[i];
    }
  }
}

__global__ __launch_bounds__(256)
void attn_pass2_kernel(const float* __restrict__ qkv,
                       const float* __restrict__ mrow,
                       const float* __restrict__ lrow,
                       float* __restrict__ ctx,
                       float* __restrict__ colsum, int N) {
  __shared__ __bf16 pbuf[8 * 512];            // per-wave 16x32 P tile (bf16)
  const int wv   = threadIdx.x >> 5;
  const int wave = (blockIdx.x << 3) + wv;
  const int lane = threadIdx.x & 31;
  const int half = lane >> 4;
  const int n    = lane & 15;
  const int qpt  = N >> 4;
  const int h    = wave / qpt;
  const int q0   = (wave % qpt) << 4;
  __bf16* pb = pbuf + wv * 512;
  const float scale = 0.17677669529663689f;

  const v16bf aq = load_a_bf16(qkv + (size_t)(q0 + n) * 384 + h * 32, half, scale);

  float mr[8], invl[8];
#pragma unroll
  for (int i = 0; i < 8; ++i) {
    const int row = q0 + i + 8 * half;
    mr[i]   = mrow[h * N + row];
    invl[i] = 1.0f / lrow[h * N + row];
  }

  v8f acc1 = {}, acc2 = {};
  for (int j0 = 0; j0 < N; j0 += 32) {
    v16bf bk1, bk2;
    {
      const float* k1 = qkv + (size_t)(j0 + n) * 384 + 128 + h * 32 + 16 * half;
      const float* k2 = qkv + (size_t)(j0 + 16 + n) * 384 + 128 + h * 32 + 16 * half;
#pragma unroll
      for (int i = 0; i < 16; ++i) { bk1[i] = (__bf16)k1[i]; bk2[i] = (__bf16)k2[i]; }
    }
    v8f s1 = {}, s2 = {};
    s1 = __builtin_amdgcn_wmma_f32_16x16x32_bf16(false, aq, false, bk1, (short)0, s1,
                                                 false, false);
    s2 = __builtin_amdgcn_wmma_f32_16x16x32_bf16(false, aq, false, bk2, (short)0, s2,
                                                 false, false);
    float cs1 = 0.f, cs2 = 0.f;
#pragma unroll
    for (int i = 0; i < 8; ++i) {
      const float p1 = __expf(s1[i] - mr[i]) * invl[i];
      const float p2 = __expf(s2[i] - mr[i]) * invl[i];
      cs1 += p1; cs2 += p2;
      const int row = i + 8 * half;           // C layout row within tile
      pb[row * 32 + n]      = (__bf16)p1;
      pb[row * 32 + 16 + n] = (__bf16)p2;
    }
    atomicAdd(&colsum[j0 + n],      cs1);
    atomicAdd(&colsum[j0 + 16 + n], cs2);

    asm volatile("" ::: "memory");  // keep DS stores before DS loads (in-order per wave)

    // reload P as the A-operand (16x32, key dim = K)
    v16bf ap;
#pragma unroll
    for (int i = 0; i < 8; ++i) {
      ap[i]     = pb[n * 32 + 16 * half + i];
      ap[8 + i] = pb[n * 32 + 16 * half + 8 + i];
    }
    // V tile as B operand: col = output dim, K = key within tile
    v16bf bv1, bv2;
#pragma unroll
    for (int i = 0; i < 16; ++i) {
      const float* vr = qkv + (size_t)(j0 + i + 16 * half) * 384 + 256 + h * 32;
      bv1[i] = (__bf16)vr[n];
      bv2[i] = (__bf16)vr[16 + n];
    }
    acc1 = __builtin_amdgcn_wmma_f32_16x16x32_bf16(false, ap, false, bv1, (short)0,
                                                   acc1, false, false);
    acc2 = __builtin_amdgcn_wmma_f32_16x16x32_bf16(false, ap, false, bv2, (short)0,
                                                   acc2, false, false);
  }
#pragma unroll
  for (int i = 0; i < 8; ++i) {
    const int row = q0 + i + 8 * half;
    ctx[(size_t)row * 128 + h * 32 + n]      = acc1[i];
    ctx[(size_t)row * 128 + h * 32 + 16 + n] = acc2[i];
  }
}

// ---------------- tail ----------------

__global__ __launch_bounds__(256)
void ni_kernel(const float* __restrict__ cs, float* __restrict__ out) {
  const int i = blockIdx.x * 256 + threadIdx.x;
  out[1 + i] = cs[i] * (1.f / (4.f * (float)N_NODES));
}

__global__ __launch_bounds__(256)
void graph_repr_kernel(const float* __restrict__ att,
                       const float* __restrict__ ni,
                       float* __restrict__ gr, int N) {
  const int d = blockIdx.x;                  // 128 dims
  float s = 0.f;
  for (int i = threadIdx.x; i < N; i += 256)
    s += att[(size_t)i * 128 + d] * ni[i];
  __shared__ float red[256];
  red[threadIdx.x] = s; __syncthreads();
  for (int off = 128; off >= 1; off >>= 1) {
    if (threadIdx.x < off) red[threadIdx.x] += red[threadIdx.x + off];
    __syncthreads();
  }
  if (threadIdx.x == 0) gr[d] = red[0] / (float)N;
}

__global__ __launch_bounds__(64)
void score_kernel(const float* __restrict__ gr,
                  const float* __restrict__ W1, const float* __restrict__ b1,
                  const float* __restrict__ W2, const float* __restrict__ b2,
                  float* __restrict__ out) {
  const int t = threadIdx.x;                 // 64 threads
  float acc = b1[t];
  for (int d = 0; d < 128; ++d) acc += gr[d] * W1[d * 64 + t];
  acc = fmaxf(acc, 0.f);
  __shared__ float red[64];
  red[t] = acc * W2[t];
  __syncthreads();
  if (t == 0) {
    float s = b2[0];
    for (int i = 0; i < 64; ++i) s += red[i];
    out[0] = 1.f / (1.f + __expf(-s));
  }
}

// ---------------- host ----------------

extern "C" void kernel_launch(void* const* d_in, const int* in_sizes, int n_in,
                              void* d_out, int out_size, void* d_ws, size_t ws_size,
                              hipStream_t stream) {
  const int N = N_NODES;
  const float* x        = (const float*)d_in[0];
  const int*   ei       = (const int*)  d_in[1];
  const float* W_in     = (const float*)d_in[2];
  const float* b_in     = (const float*)d_in[3];
  const float* gat_W0   = (const float*)d_in[4];
  const float* gat_W12  = (const float*)d_in[5];
  const float* att_src  = (const float*)d_in[6];   // [3,4,128]
  const float* att_dst  = (const float*)d_in[7];
  const float* gat_b    = (const float*)d_in[8];   // [3,512]
  const float* W_final  = (const float*)d_in[9];
  const float* b_final  = (const float*)d_in[10];
  const float* mha_in_w = (const float*)d_in[11];  // [384,128]
  const float* mha_in_b = (const float*)d_in[12];
  const float* mha_ow   = (const float*)d_in[13];  // [128,128]
  const float* mha_ob   = (const float*)d_in[14];
  const float* W1       = (const float*)d_in[15];
  const float* b1       = (const float*)d_in[16];
  const float* W2       = (const float*)d_in[17];
  const float* b2       = (const float*)d_in[18];
  float* out = (float*)d_out;

  // workspace layout (floats); peak ~56.6 MB
  float* ws = (float*)d_ws;
  float* h0    = ws;                          // 8192*128   (reused: nf)
  float* hl    = ws + 1048576;                // 8192*512   (reused: qkv)
  float* agg   = ws + 5242880;                // 8192*512   (reused: ctx)
  float* hA    = ws + 9437184;                // 8192*512   (reused: attended)
  float* als   = ws + 13631488;               // 8192*4
  float* ald   = ws + 13664256;               // 8192*4
  float* smax  = ws + 13697024;               // 8192*4
  float* ssum  = ws + 13729792;               // 8192*4
  float* ev    = ws + 13762560;               // E_TOT*4 = 1081344
  float* mrow  = ev;                          // reuse after GAT: 4*8192
  float* lrow  = ev + 32768;                  // 4*8192
  float* csum  = ev + 65536;                  // 8192
  float* gr    = ev + 73728;                  // 128
  float* nf    = h0;
  float* qkv   = hl;
  float* ctx   = agg;
  float* attnd = hA;

  // h0 = relu(x @ W_in + b_in)                          [8192,256]x[256,128]
  gemm_f32_kernel<0, 1, 1><<<dim3(128 / 64, N / 128), 256, 0, stream>>>(
      x, W_in, b_in, h0, N, 128, 256);

  const float* feat = h0;
  int Kf = 128;
  for (int L = 0; L < 3; ++L) {
    const float* W = (L == 0) ? gat_W0 : (gat_W12 + (size_t)(L - 1) * HD * HD);
    gemm_f32_kernel<0, 0, 0><<<dim3(HD / 64, N / 128), 256, 0, stream>>>(
        feat, W, nullptr, hl, N, HD, Kf);
    gat_alpha_kernel<<<(N * 4) / 8, 256, 0, stream>>>(
        hl, att_src + L * 512, att_dst + L * 512, als, ald);
    fill_kernel<<<(N * 4 + 255) / 256, 256, 0, stream>>>(smax, -3.4e38f, N * 4);
    fill_kernel<<<(N * 4 + 255) / 256, 256, 0, stream>>>(ssum, 0.f, N * 4);
    fill_kernel<<<(N * HD + 255) / 256, 256, 0, stream>>>(agg, 0.f, N * HD);
    edge_score_kernel<<<(E_TOT * 4 + 255) / 256, 256, 0, stream>>>(ei, als, ald, ev, smax);
    edge_exp_kernel<<<(E_TOT * 4 + 255) / 256, 256, 0, stream>>>(ei, ev, smax, ssum);
    edge_agg_kernel<<<(E_TOT * 4 + 7) / 8, 256, 0, stream>>>(ei, ev, ssum, hl, agg);
    bias_relu_kernel<<<(N * HD + 255) / 256, 256, 0, stream>>>(agg, gat_b + L * HD, hA,
                                                               N * HD);
    feat = hA; Kf = HD;
  }

  // nf = hA @ W_final + b_final ; qkv = nf @ mha_in_w^T + mha_in_b
  gemm_f32_kernel<0, 0, 1><<<dim3(128 / 64, N / 128), 256, 0, stream>>>(
      hA, W_final, b_final, nf, N, 128, HD);
  gemm_f32_kernel<1, 0, 1><<<dim3(384 / 64, N / 128), 256, 0, stream>>>(
      nf, mha_in_w, mha_in_b, qkv, N, 384, 128);

  // flash attention (two passes)
  attn_pass1_kernel<<<(4 * (N / 16)) / 8, 256, 0, stream>>>(qkv, mrow, lrow, N);
  fill_kernel<<<(N + 255) / 256, 256, 0, stream>>>(csum, 0.f, N);
  attn_pass2_kernel<<<(4 * (N / 16)) / 8, 256, 0, stream>>>(qkv, mrow, lrow, ctx, csum, N);

  // attended = ctx @ mha_out_w^T + mha_out_b
  gemm_f32_kernel<1, 0, 1><<<dim3(128 / 64, N / 128), 256, 0, stream>>>(
      ctx, mha_ow, mha_ob, attnd, N, 128, 128);

  // node importances -> out[1..N]; graph repr; final score -> out[0]
  ni_kernel<<<N / 256, 256, 0, stream>>>(csum, out);
  graph_repr_kernel<<<128, 256, 0, stream>>>(attnd, out + 1, gr, N);
  score_kernel<<<1, 64, 0, stream>>>(gr, W1, b1, W2, b2, out);
}
